// GLSTMCell_76879914598610
// MI455X (gfx1250) — matrix operand order repaired and checked by reading, not compiled, not run on hardware
//
#include <hip/hip_runtime.h>
#include <hip/hip_bf16.h>

// ---------------------------------------------------------------------------
// Graph-LSTM GNN for MI455X (gfx1250, wave32, WMMA).
//
// d_in flattening assumption (setup_inputs dict insertion order, recursive):
//  0 enc.W_in[64,16]   1 enc.b_in[64]   2 enc.W_h0[64,64]  3 enc.W_h1[64,64]
//  4 enc.b_h0[64]      5 enc.b_h1[64]   6 enc.W_out[32,64] 7 enc.b_out[32]
//  8 enc.gamma[32]     9 enc.beta[32]
// 10 dec.W_in[64,32]  11 dec.b_in[64]  12 dec.W_h0[64,64] 13 dec.W_h1[64,64]
// 14 dec.b_h0[64]     15 dec.b_h1[64]  16 dec.W_out[2,64] 17 dec.b_out[2]
// 18 Wi[32,32] 19 bi[32] 20 Ui[32,32] 21 Wf[32,5] 22 bf[32] 23 Uf[32,32]
// 24 Wo[32,32] 25 bo[32] 26 Uo[32,32] 27 Wu[32,32] 28 bu[32] 29 Uu[32,32]
// 30 nfeat[N,16] 31 efeat[E,5] 32 h0[N,32] 33 c0[N,32] 34 src[E] 35 dst[E]
// ---------------------------------------------------------------------------

typedef __attribute__((ext_vector_type(16))) _Float16 v16h;
typedef __attribute__((ext_vector_type(8)))  _Float16 h8v;
typedef __attribute__((ext_vector_type(8)))  float    v8f;
typedef __attribute__((ext_vector_type(4)))  float    f4v;

#define WMMA_F16(a, b, c) \
  __builtin_amdgcn_wmma_f32_16x16x32_f16(false, (a), false, (b), (short)0, (c), false, false)

__device__ __forceinline__ float sigmoidf_(float x) { return 1.0f / (1.0f + __expf(-x)); }
__device__ __forceinline__ void  lds_fence() { asm volatile("s_wait_dscnt 0x0" ::: "memory"); }
__device__ __forceinline__ void  atomic_add_f32(float* p, float v) {
  __hip_atomic_fetch_add(p, v, __ATOMIC_RELAXED, __HIP_MEMORY_SCOPE_AGENT);
}

// Concatenate two 8-wide f16 runs into a WMMA A/B fragment register block.
__device__ __forceinline__ v16h make_frag(h8v u, h8v v) {
  v16h a;
#pragma unroll
  for (int i = 0; i < 8; ++i) { a[i] = u[i]; a[8 + i] = v[i]; }
  return a;
}

// A-fragment (16-bit A 16x32, ISA 7.12.2) from f16 LDS act[16][ld].
// Lane m=lane&15 row; runs at k0+8*half and k0+16+8*half (16B aligned, ld%16==0).
__device__ __forceinline__ v16h load_A_lds16(const _Float16* act, int ld, int k0,
                                             int lane) {
  const int m = lane & 15, half = lane >> 4;
  const _Float16* rp = act + m * ld + k0 + 8 * half;
  h8v u = *(const h8v*)(rp);
  h8v v = *(const h8v*)(rp + 16);
  return make_frag(u, v);
}

// B-fragment: W row-major [outF][ldk] f16 in LDS; runs at k0+8*half and
// k0+16+8*half (same k order as the A fragment).
__device__ __forceinline__ v16h load_B_lds(const _Float16* w, int ldk, int n0,
                                           int k0, int lane) {
  const int n = n0 + (lane & 15), half = lane >> 4;
  const _Float16* rp = w + n * ldk + k0 + 8 * half;
  h8v lo = *(const h8v*)(rp);
  h8v hi = *(const h8v*)(rp + 16);
  return make_frag(lo, hi);
}

// A-fragment directly from a global f32 row of 32 floats (16B aligned):
// two float4 pairs per lane-half, converted to f16.
__device__ __forceinline__ v16h load_A_row_f32(const float* rowp, int half,
                                               f4v keep[4]) {
  const f4v* p = (const f4v*)rowp;
  f4v v0 = p[2 * half], v1 = p[2 * half + 1];
  f4v v2 = p[4 + 2 * half], v3 = p[4 + 2 * half + 1];
  keep[0] = v0; keep[1] = v1; keep[2] = v2; keep[3] = v3;
  v16h a;
#pragma unroll
  for (int i = 0; i < 4; ++i) {
    a[i]      = (_Float16)v0[i];
    a[4 + i]  = (_Float16)v1[i];
    a[8 + i]  = (_Float16)v2[i];
    a[12 + i] = (_Float16)v3[i];
  }
  return a;
}

// dst[16][outF] = act(src[16][inF] @ W.T + b). inF, outF multiples of 16;
// inF multiple of 32 (zero-padded by the caller where needed).
// ACT: 0 = identity, 1 = leaky_relu(0.01).
template <int ACT>
__device__ __forceinline__ void wave_layer16(const _Float16* src, int inF,
                                             _Float16* dst, int outF,
                                             const _Float16* w, const float* b,
                                             int lane) {
  const int half = lane >> 4;
  for (int n0 = 0; n0 < outF; n0 += 16) {
    v8f acc = {};
    for (int k0 = 0; k0 < inF; k0 += 32) {
      v16h a  = load_A_lds16(src, inF, k0, lane);
      v16h bb = load_B_lds(w, inF, n0, k0, lane);
      acc = WMMA_F16(a, bb, acc);
    }
    const int n = n0 + (lane & 15);
    const float bias = b[n];
#pragma unroll
    for (int r = 0; r < 8; ++r) {
      float v = acc[r] + bias;
      if (ACT) v = (v > 0.0f) ? v : 0.01f * v;
      dst[(r + 8 * half) * outF + n] = (_Float16)v;  // C layout: M = r + 8*half
    }
  }
  lds_fence();
}

__device__ __forceinline__ void stage_w16(_Float16* d, const float* s, int n,
                                          int tid, int nt) {
  for (int i = tid; i < n; i += nt) d[i] = (_Float16)s[i];
}
__device__ __forceinline__ void stage_f32(float* d, const float* s, int n,
                                          int tid, int nt) {
  for (int i = tid; i < n; i += nt) d[i] = s[i];
}

// --------------------------- encode + gate-precompute ----------------------
__global__ __launch_bounds__(256) void encode_kernel(
    const float* __restrict__ nfeat,
    const float* W_in, const float* b_in, const float* W_h0, const float* W_h1,
    const float* b_h0, const float* b_h1, const float* W_out, const float* b_out,
    const float* gamma, const float* beta,
    const float* Wi, const float* bi, const float* Wo, const float* bo,
    const float* Wu, const float* bu,
    float* __restrict__ Wi_x, float* __restrict__ Wo_x, float* __restrict__ Wu_x,
    int N) {
  __shared__ __align__(16) _Float16 wIn[64 * 32];  // K zero-padded 16 -> 32
  __shared__ __align__(16) _Float16 wH0[64 * 64], wH1[64 * 64], wOut[32 * 64];
  __shared__ __align__(16) _Float16 wWi[32 * 32], wWo[32 * 32], wWu[32 * 32];
  __shared__ __align__(16) float bIn[64], bH0[64], bH1[64];
  __shared__ __align__(16) float bOutS[32], gam[32], bet[32], biS[32], boS[32], buS[32];
  __shared__ __align__(16) _Float16 bufA[8][16 * 64];
  __shared__ __align__(16) _Float16 bufB[8][16 * 64];

  const int tid = threadIdx.x;
  for (int i = tid; i < 64 * 32; i += 256) {  // stage W_in with zero K-pad
    int o = i >> 5, k = i & 31;
    wIn[i] = (k < 16) ? (_Float16)W_in[o * 16 + k] : (_Float16)0.0f;
  }
  stage_w16(wH0, W_h0, 64 * 64, tid, 256);
  stage_w16(wH1, W_h1, 64 * 64, tid, 256);
  stage_w16(wOut, W_out, 32 * 64, tid, 256);
  stage_w16(wWi, Wi, 32 * 32, tid, 256);
  stage_w16(wWo, Wo, 32 * 32, tid, 256);
  stage_w16(wWu, Wu, 32 * 32, tid, 256);
  stage_f32(bIn, b_in, 64, tid, 256);
  stage_f32(bH0, b_h0, 64, tid, 256);
  stage_f32(bH1, b_h1, 64, tid, 256);
  stage_f32(bOutS, b_out, 32, tid, 256);
  stage_f32(gam, gamma, 32, tid, 256);
  stage_f32(bet, beta, 32, tid, 256);
  stage_f32(biS, bi, 32, tid, 256);
  stage_f32(boS, bo, 32, tid, 256);
  stage_f32(buS, bu, 32, tid, 256);
  __syncthreads();

  const int wave = tid >> 5, lane = tid & 31;
  const int half = lane >> 4, nl = lane & 15;
  const long base = ((long)blockIdx.x * 8 + wave) * 16;
  _Float16* A = bufA[wave];
  _Float16* B = bufB[wave];

  // Stage input tile [16 nodes][32] (cols 16..31 zero), clamped rows.
  for (int i = lane; i < 16 * 32; i += 32) {
    int m = i >> 5, k = i & 31;
    long row = base + m; if (row >= N) row = N - 1;
    A[i] = (k < 16) ? (_Float16)nfeat[row * 16 + k] : (_Float16)0.0f;
  }
  lds_fence();

  wave_layer16<1>(A, 32, B, 64, wIn, bIn, lane);
  wave_layer16<1>(B, 64, A, 64, wH0, bH0, lane);
  wave_layer16<1>(A, 64, B, 64, wH1, bH1, lane);
  wave_layer16<0>(B, 64, A, 32, wOut, bOutS, lane);

  // LayerNorm over 32 features per row (lanes 0-15 own a row; vectorized).
  if (lane < 16) {
    h8v* rp = (h8v*)(A + lane * 32);
    h8v x0 = rp[0], x1 = rp[1], x2 = rp[2], x3 = rp[3];
    float xs[32];
#pragma unroll
    for (int i = 0; i < 8; ++i) {
      xs[i] = (float)x0[i]; xs[8 + i] = (float)x1[i];
      xs[16 + i] = (float)x2[i]; xs[24 + i] = (float)x3[i];
    }
    float mu = 0.0f;
#pragma unroll
    for (int i = 0; i < 32; ++i) mu += xs[i];
    mu *= (1.0f / 32.0f);
    float var = 0.0f;
#pragma unroll
    for (int i = 0; i < 32; ++i) { float d = xs[i] - mu; var += d * d; }
    var *= (1.0f / 32.0f);
    const float inv = rsqrtf(var + 1e-5f);
#pragma unroll
    for (int i = 0; i < 32; ++i) xs[i] = (xs[i] - mu) * inv * gam[i] + bet[i];
#pragma unroll
    for (int i = 0; i < 8; ++i) {
      x0[i] = (_Float16)xs[i];       x1[i] = (_Float16)xs[8 + i];
      x2[i] = (_Float16)xs[16 + i];  x3[i] = (_Float16)xs[24 + i];
    }
    rp[0] = x0; rp[1] = x1; rp[2] = x2; rp[3] = x3;
  }
  lds_fence();

  // Gate preactivations: Wi_x/Wo_x/Wu_x = proc_n @ W*.T + b*
  const v16h aP = load_A_lds16(A, 32, 0, lane);
  auto emit_gate = [&](const _Float16* w, const float* bb, float* outg) {
    for (int n0 = 0; n0 < 32; n0 += 16) {
      v16h bf_ = load_B_lds(w, 32, n0, 0, lane);
      v8f acc = {};
      acc = WMMA_F16(aP, bf_, acc);
      const int n = n0 + nl;
      const float bias = bb[n];
#pragma unroll
      for (int r = 0; r < 8; ++r) {
        long row = base + r + 8 * half; if (row >= N) row = N - 1;
        outg[row * 32 + n] = acc[r] + bias;
      }
    }
  };
  emit_gate(wWi, biS, Wi_x);
  emit_gate(wWo, boS, Wo_x);
  emit_gate(wWu, buS, Wu_x);
}

// ------------------------------- edge pass ---------------------------------
// 16 edges per wave. h/c/hsum/fcsum are L2-resident (12.8 MB each in 192 MB L2);
// Wf_e recomputed inline instead of streaming 205 MB/iter from HBM.
__global__ __launch_bounds__(256) void edge_kernel(
    const float* __restrict__ h, const float* __restrict__ c,
    const float* __restrict__ efeat,
    const int* __restrict__ src, const int* __restrict__ dst,
    const float* Uf, const float* Wf, const float* bf,
    float* __restrict__ hsum, float* __restrict__ fcsum) {
  __shared__ __align__(16) _Float16 uf16[32 * 32];
  __shared__ __align__(16) float wfS[32 * 5], bfS[32];
  const int tid = threadIdx.x;
  stage_w16(uf16, Uf, 32 * 32, tid, 256);
  stage_f32(wfS, Wf, 32 * 5, tid, 256);
  stage_f32(bfS, bf, 32, tid, 256);
  __syncthreads();

  const int wave = tid >> 5, lane = tid & 31;
  const int half = lane >> 4, nl = lane & 15;
  const long base = ((long)blockIdx.x * 8 + wave) * 16;  // grid covers full tiles

  // Gather h[src] row as A fragment (two float4 pairs per lane-half).
  const long eA = base + nl;
  const int sA = src[eA];
  f4v hv[4];
  const v16h a = load_A_row_f32(h + (long)sA * 32, half, hv);

  // segment_sum(h_src, dst): lane halves cover disjoint K sets {0..7,16..23}
  // vs {8..15,24..31}.
  const int dA = dst[eA];
  float* hs = hsum + (long)dA * 32;
  const int kb1 = 8 * half, kb2 = 16 + 8 * half;
#pragma unroll
  for (int i = 0; i < 4; ++i) {
    atomic_add_f32(&hs[kb1 + i],     hv[0][i]);
    atomic_add_f32(&hs[kb1 + 4 + i], hv[1][i]);
    atomic_add_f32(&hs[kb2 + i],     hv[2][i]);
    atomic_add_f32(&hs[kb2 + 4 + i], hv[3][i]);
  }

  // Hoist per-edge C-row metadata (reused across both N-tiles).
  int s2a[8], d2a[8];
  float efa[8][5];
#pragma unroll
  for (int r = 0; r < 8; ++r) {
    const long e2 = base + r + 8 * half;  // C layout row M = r + 8*half
    s2a[r] = src[e2];
    d2a[r] = dst[e2];
    const float* ef = efeat + e2 * 5;
#pragma unroll
    for (int q = 0; q < 5; ++q) efa[r][q] = ef[q];
  }

  // f = sigmoid(Wf_e + h_src @ Uf.T); fc = f * sigmoid(c[src]); scatter-add.
  for (int n0 = 0; n0 < 32; n0 += 16) {
    v16h bfr = load_B_lds(uf16, 32, n0, 0, lane);
    v8f acc = {};
    acc = WMMA_F16(a, bfr, acc);
    const int n = n0 + nl;
    float wfr[5];
#pragma unroll
    for (int q = 0; q < 5; ++q) wfr[q] = wfS[n * 5 + q];
    const float bfn = bfS[n];
#pragma unroll
    for (int r = 0; r < 8; ++r) {
      float pre = bfn;
#pragma unroll
      for (int q = 0; q < 5; ++q) pre += efa[r][q] * wfr[q];
      const float f = sigmoidf_(pre + acc[r]);
      const float fc = f * sigmoidf_(c[(long)s2a[r] * 32 + n]);
      atomic_add_f32(&fcsum[(long)d2a[r] * 32 + n], fc);
    }
  }
}

// Scalar tail for E not divisible by 128 (one thread per edge-feature).
__global__ void edge_tail_kernel(const float* h, const float* c, const float* efeat,
                                 const int* src, const int* dst, const float* Uf,
                                 const float* Wf, const float* bf, float* hsum,
                                 float* fcsum, long e0, long E) {
  long t = (long)blockIdx.x * blockDim.x + threadIdx.x;
  long e = e0 + (t >> 5);
  int n = (int)(t & 31);
  if (e >= E) return;
  int s = src[e], d = dst[e];
  float uf = 0.0f;
  for (int k = 0; k < 32; ++k) uf += h[(long)s * 32 + k] * Uf[n * 32 + k];
  float pre = bf[n];
  for (int q = 0; q < 5; ++q) pre += efeat[e * 5 + q] * Wf[n * 5 + q];
  float f = sigmoidf_(pre + uf);
  atomic_add_f32(&fcsum[(long)d * 32 + n], f * sigmoidf_(c[(long)s * 32 + n]));
  atomic_add_f32(&hsum[(long)d * 32 + n], h[(long)s * 32 + n]);
}

// ------------------------------- node pass ---------------------------------
__global__ __launch_bounds__(256) void node_kernel(
    const float* __restrict__ hsum, const float* __restrict__ fcsum,
    const float* __restrict__ Wi_x, const float* __restrict__ Wo_x,
    const float* __restrict__ Wu_x,
    const float* Ui, const float* Uo, const float* Uu,
    float* __restrict__ h, float* __restrict__ c, int N) {
  __shared__ __align__(16) _Float16 ui16[32 * 32], uo16[32 * 32], uu16[32 * 32];
  const int tid = threadIdx.x;
  stage_w16(ui16, Ui, 32 * 32, tid, 256);
  stage_w16(uo16, Uo, 32 * 32, tid, 256);
  stage_w16(uu16, Uu, 32 * 32, tid, 256);
  __syncthreads();

  const int wave = tid >> 5, lane = tid & 31;
  const int half = lane >> 4, nl = lane & 15;
  const long base = ((long)blockIdx.x * 8 + wave) * 16;

  long rowA = base + nl; if (rowA >= N) rowA = N - 1;
  f4v dummy[4];
  const v16h a = load_A_row_f32(hsum + rowA * 32, half, dummy);

  for (int n0 = 0; n0 < 32; n0 += 16) {
    v16h bi_ = load_B_lds(ui16, 32, n0, 0, lane);
    v16h bo_ = load_B_lds(uo16, 32, n0, 0, lane);
    v16h bu_ = load_B_lds(uu16, 32, n0, 0, lane);
    v8f ai = {}, ao = {}, au = {};
    ai = WMMA_F16(a, bi_, ai);
    ao = WMMA_F16(a, bo_, ao);
    au = WMMA_F16(a, bu_, au);
    const int n = n0 + nl;
#pragma unroll
    for (int r = 0; r < 8; ++r) {
      long row = base + r + 8 * half; if (row >= N) row = N - 1;
      const long idx = row * 32 + n;
      const float iv = sigmoidf_(Wi_x[idx] + ai[r]);
      const float ov = sigmoidf_(Wo_x[idx] + ao[r]);
      const float uv = tanhf(Wu_x[idx] + au[r]);
      const float cn = fcsum[idx] + iv * uv;
      c[idx] = cn;
      h[idx] = ov * tanhf(cn);
    }
  }
}

// --------------------------------- decode ----------------------------------
__global__ __launch_bounds__(256) void decode_kernel(
    const float* __restrict__ h,
    const float* W_in, const float* b_in, const float* W_h0, const float* W_h1,
    const float* b_h0, const float* b_h1, const float* W_out, const float* b_out,
    float* __restrict__ out, int N) {
  __shared__ __align__(16) _Float16 wIn[64 * 32], wH0[64 * 64], wH1[64 * 64];
  __shared__ __align__(16) float wOutD[2 * 64], bIn[64], bH0[64], bH1[64], bOutD[2];
  __shared__ __align__(16) _Float16 bufA[8][16 * 64];
  __shared__ __align__(16) _Float16 bufB[8][16 * 64];

  const int tid = threadIdx.x;
  stage_w16(wIn, W_in, 64 * 32, tid, 256);
  stage_w16(wH0, W_h0, 64 * 64, tid, 256);
  stage_w16(wH1, W_h1, 64 * 64, tid, 256);
  stage_f32(wOutD, W_out, 2 * 64, tid, 256);
  stage_f32(bIn, b_in, 64, tid, 256);
  stage_f32(bH0, b_h0, 64, tid, 256);
  stage_f32(bH1, b_h1, 64, tid, 256);
  stage_f32(bOutD, b_out, 2, tid, 256);
  __syncthreads();

  const int wave = tid >> 5, lane = tid & 31;
  const long base = ((long)blockIdx.x * 8 + wave) * 16;
  _Float16* A = bufA[wave];
  _Float16* B = bufB[wave];

  for (int i = lane; i < 16 * 32; i += 32) {
    int m = i >> 5, k = i & 31;
    long row = base + m; if (row >= N) row = N - 1;
    A[m * 32 + k] = (_Float16)h[row * 32 + k];
  }
  lds_fence();

  wave_layer16<1>(A, 32, B, 64, wIn, bIn, lane);
  wave_layer16<1>(B, 64, A, 64, wH0, bH0, lane);
  wave_layer16<1>(A, 64, B, 64, wH1, bH1, lane);

  // Final 64 -> 2: one (node, out) pair per lane, vectorized LDS dot.
  const int m = lane >> 1, oo = lane & 1;
  float acc = bOutD[oo];
  const h8v* rp = (const h8v*)(B + m * 64);
#pragma unroll
  for (int t = 0; t < 8; ++t) {
    h8v x = rp[t];
#pragma unroll
    for (int i = 0; i < 8; ++i) acc += (float)x[i] * wOutD[oo * 64 + 8 * t + i];
  }
  long row = base + m; if (row >= N) row = N - 1;
  out[row * 2 + oo] = acc;
}

// ------------------------------ init helpers -------------------------------
__global__ void copy2_kernel(const float* a0, const float* b0, float* a, float* b,
                             long n) {
  long i = (long)blockIdx.x * blockDim.x + threadIdx.x;
  if (i < n) { a[i] = a0[i]; b[i] = b0[i]; }
}
__global__ void zero2_kernel(float* a, float* b, long n) {
  long i = (long)blockIdx.x * blockDim.x + threadIdx.x;
  if (i < n) { a[i] = 0.0f; b[i] = 0.0f; }
}

// ------------------------------ host launch --------------------------------
extern "C" void kernel_launch(void* const* d_in, const int* in_sizes, int n_in,
                              void* d_out, int out_size, void* d_ws, size_t ws_size,
                              hipStream_t stream) {
  const float* eWin  = (const float*)d_in[0];
  const float* eBin  = (const float*)d_in[1];
  const float* eWh0  = (const float*)d_in[2];
  const float* eWh1  = (const float*)d_in[3];
  const float* eBh0  = (const float*)d_in[4];
  const float* eBh1  = (const float*)d_in[5];
  const float* eWout = (const float*)d_in[6];
  const float* eBout = (const float*)d_in[7];
  const float* eGam  = (const float*)d_in[8];
  const float* eBet  = (const float*)d_in[9];
  const float* dWin  = (const float*)d_in[10];
  const float* dBin  = (const float*)d_in[11];
  const float* dWh0  = (const float*)d_in[12];
  const float* dWh1  = (const float*)d_in[13];
  const float* dBh0  = (const float*)d_in[14];
  const float* dBh1  = (const float*)d_in[15];
  const float* dWout = (const float*)d_in[16];
  const float* dBout = (const float*)d_in[17];
  const float* Wi = (const float*)d_in[18];
  const float* bi = (const float*)d_in[19];
  const float* Ui = (const float*)d_in[20];
  const float* Wf = (const float*)d_in[21];
  const float* bf = (const float*)d_in[22];
  const float* Uf = (const float*)d_in[23];
  const float* Wo = (const float*)d_in[24];
  const float* bo = (const float*)d_in[25];
  const float* Uo = (const float*)d_in[26];
  const float* Wu = (const float*)d_in[27];
  const float* bu = (const float*)d_in[28];
  const float* Uu = (const float*)d_in[29];
  const float* nfeat = (const float*)d_in[30];
  const float* efeat = (const float*)d_in[31];
  const float* h0 = (const float*)d_in[32];
  const float* c0 = (const float*)d_in[33];
  const int* src = (const int*)d_in[34];
  const int* dst = (const int*)d_in[35];

  const int N = in_sizes[30] / 16;     // nfeat is [N,16]
  const long E = (long)in_sizes[34];   // src is [E]

  float* ws = (float*)d_ws;
  const size_t nh = (size_t)N * 32;
  float* h     = ws + 0 * nh;
  float* c     = ws + 1 * nh;
  float* hsum  = ws + 2 * nh;
  float* fcsum = ws + 3 * nh;
  float* Wi_x  = ws + 4 * nh;
  float* Wo_x  = ws + 5 * nh;
  float* Wu_x  = ws + 6 * nh;

  const int nodeBlocks = (N + 127) / 128;          // 8 waves x 16 nodes per block
  const long tot = (long)N * 32;
  const int cpyBlocks = (int)((tot + 255) / 256);
  const long Emain = (E / 128) * 128;
  const int edgeBlocks = (int)(Emain / 128);
  const long Etail = E - Emain;

  encode_kernel<<<nodeBlocks, 256, 0, stream>>>(
      nfeat, eWin, eBin, eWh0, eWh1, eBh0, eBh1, eWout, eBout, eGam, eBet,
      Wi, bi, Wo, bo, Wu, bu, Wi_x, Wo_x, Wu_x, N);
  copy2_kernel<<<cpyBlocks, 256, 0, stream>>>(h0, c0, h, c, tot);

  for (int it = 0; it < 3; ++it) {
    zero2_kernel<<<cpyBlocks, 256, 0, stream>>>(hsum, fcsum, tot);
    if (edgeBlocks > 0)
      edge_kernel<<<edgeBlocks, 256, 0, stream>>>(h, c, efeat, src, dst, Uf, Wf,
                                                  bf, hsum, fcsum);
    if (Etail > 0) {
      long tthreads = Etail * 32;
      edge_tail_kernel<<<(int)((tthreads + 255) / 256), 256, 0, stream>>>(
          h, c, efeat, src, dst, Uf, Wf, bf, hsum, fcsum, Emain, E);
    }
    node_kernel<<<nodeBlocks, 256, 0, stream>>>(hsum, fcsum, Wi_x, Wo_x, Wu_x,
                                                Ui, Uo, Uu, h, c, N);
  }

  decode_kernel<<<nodeBlocks, 256, 0, stream>>>(
      h, dWin, dBin, dWh0, dWh1, dBh0, dBh1, dWout, dBout, (float*)d_out, N);
}
